// GAE_43447889166448
// MI455X (gfx1250) — matrix operand order repaired and checked by reading, not compiled, not run on hardware
//
#include <hip/hip_runtime.h>

// ---------------------------------------------------------------------------
// GCN autoencoder forward for MI455X (gfx1250, wave32, WMMA).
//
//   * agg(X) @ W == agg(X @ W)  -> GEMM before gather (smaller gather dim).
//   * fp32 GEMMs as split-bf16 (hi+lo): Ahi*Bhi + Ahi*Blo + Alo*Bhi with
//     v_wmma_f32_16x16x32_bf16, fp32 accumulate (~fp32 accuracy, 8x the
//     MACs/instr of the f32 16x16x4 path).
//   * z@z.T dominates: 256MB store (~11us HBM floor at 23.3TB/s); Z operands
//     (2+2MB) stay hot in 192MB L2.
//   * 64x32 per-wave register tile (24 WMMA : 24 b128 loads per k-step) and
//     explicit double-buffered fragments so next-step loads never WAR the
//     current step's WMMA sources (kills the 4x v_nop hazard stalls).
// ---------------------------------------------------------------------------

typedef __bf16 bf16_t;
typedef bf16_t v16bf __attribute__((ext_vector_type(16)));
typedef float  v8f   __attribute__((ext_vector_type(8)));
typedef float  v4f   __attribute__((ext_vector_type(4)));

#define NN      8192
#define EE      (8192 * 32)
#define IN_DIM  512
#define H0D     256
#define H1D     128

// ---- bf16 split helpers (bit-exact, no scalar __bf16 arithmetic needed) ----
__device__ __forceinline__ unsigned short f2bf_rn(float x) {
  unsigned int u = __float_as_uint(x);
  unsigned int r = u + 0x7FFFu + ((u >> 16) & 1u);   // round-to-nearest-even
  return (unsigned short)(r >> 16);
}
__device__ __forceinline__ float bf2f(unsigned short h) {
  return __uint_as_float(((unsigned int)h) << 16);
}

// ---- elementwise kernels ---------------------------------------------------
__global__ __launch_bounds__(256) void k_split(const float* __restrict__ x,
                                               unsigned short* __restrict__ hi,
                                               unsigned short* __restrict__ lo,
                                               int total) {
  int i = blockIdx.x * blockDim.x + threadIdx.x;
  if (i >= total) return;
  float v = x[i];
  unsigned short h = f2bf_rn(v);
  hi[i] = h;
  lo[i] = f2bf_rn(v - bf2f(h));
}

// W[K][N] fp32 -> Wt_hi/lo[N][K] bf16 (transpose so B fragments load contiguously)
__global__ __launch_bounds__(256) void k_split_t(const float* __restrict__ w,
                                                 unsigned short* __restrict__ hi,
                                                 unsigned short* __restrict__ lo,
                                                 int K, int Nc) {
  int i = blockIdx.x * blockDim.x + threadIdx.x;
  if (i >= K * Nc) return;
  int k = i / Nc;
  int n = i - k * Nc;
  float v = w[i];
  unsigned short h = f2bf_rn(v);
  size_t o = (size_t)n * K + k;
  hi[o] = h;
  lo[o] = f2bf_rn(v - bf2f(h));
}

__global__ __launch_bounds__(256) void k_zero(float* __restrict__ p, int total) {
  int i = blockIdx.x * blockDim.x + threadIdx.x;
  if (i < total) p[i] = 0.0f;
}

// segment-sum: agg[dst[e]] += Y[src[e]]  (4 floats / thread, L2-resident table)
__global__ __launch_bounds__(256) void k_scatter(const float* __restrict__ Y,
                                                 const int* __restrict__ src,
                                                 const int* __restrict__ dst,
                                                 float* __restrict__ agg,
                                                 int C, int E) {
  int t = blockIdx.x * blockDim.x + threadIdx.x;
  int pe = C >> 2;
  int e = t / pe;
  if (e >= E) return;
  int c = (t - e * pe) << 2;
  int s = src[e];
  int d = dst[e];
  v4f val = *(const v4f*)(Y + (size_t)s * C + c);
  float* o = agg + (size_t)d * C + c;
  atomicAdd(o + 0, val[0]);
  atomicAdd(o + 1, val[1]);
  atomicAdd(o + 2, val[2]);
  atomicAdd(o + 3, val[3]);
}

// x + bias (+ optional relu) -> bf16 hi/lo split
__global__ __launch_bounds__(256) void k_bias_split(const float* __restrict__ x,
                                                    const float* __restrict__ b,
                                                    unsigned short* __restrict__ hi,
                                                    unsigned short* __restrict__ lo,
                                                    int C, int total, int relu) {
  int i = blockIdx.x * blockDim.x + threadIdx.x;
  if (i >= total) return;
  float v = x[i] + b[i & (C - 1)];
  if (relu) v = v > 0.0f ? v : 0.0f;
  unsigned short h = f2bf_rn(v);
  hi[i] = h;
  lo[i] = f2bf_rn(v - bf2f(h));
}

// ---- WMMA fragment loads ---------------------------------------------------
// A 16x32 bf16 (ISA 7.12.2): lanes 0-15 row=lane hold K {0..7,16..23};
//                            lanes 16-31 hold K {8..15,24..31}.
__device__ __forceinline__ v16bf load_fragA(const unsigned short* __restrict__ base,
                                            int row, int k0, int K,
                                            int half, int idx16) {
  const unsigned short* p = base + (size_t)(row + idx16) * K + k0 + half * 8;
  union { v16bf v; v4f f[2]; } u;
  u.f[0] = *(const v4f*)(p);        // K = k0+half*8      .. +7
  u.f[1] = *(const v4f*)(p + 16);   // K = k0+16+half*8   .. +7
  return u.v;
}

// B 32x16 bf16 (same K order as A): lanes 0-15 col=lane hold K {0..7,16..23},
// lanes 16-31 hold K {8..15,24..31}.
// "base" is B^T stored [N][K] row-major, so a column is a contiguous row.
__device__ __forceinline__ v16bf load_fragB(const unsigned short* __restrict__ base,
                                            int col, int k0, int K,
                                            int half, int idx16) {
  const unsigned short* p = base + (size_t)(col + idx16) * K + k0 + half * 8;
  union { v16bf v; v4f f[2]; } u;
  u.f[0] = *(const v4f*)(p);        // K = k0+half*8      .. +7
  u.f[1] = *(const v4f*)(p + 16);   // K = k0+16+half*8   .. +7
  return u.v;
}

__device__ __forceinline__ v8f wmma_bf16(v16bf a, v16bf b, v8f c) {
  return __builtin_amdgcn_wmma_f32_16x16x32_bf16(false, a, false, b,
                                                 (short)0, c, false, false);
}

// One k-step worth of operand fragments (12 frags = 96 VGPRs).
struct Frags {
  v16bf ah[4], al[4], bh[2], bl[2];
};

__device__ __forceinline__ void load_frags(Frags& f,
                                           const unsigned short* __restrict__ Ahi,
                                           const unsigned short* __restrict__ Alo,
                                           const unsigned short* __restrict__ Bhi,
                                           const unsigned short* __restrict__ Blo,
                                           int m0, int n0, int k0, int K,
                                           int half, int idx16) {
#pragma unroll
  for (int i = 0; i < 4; ++i) {
    f.ah[i] = load_fragA(Ahi, m0 + 16 * i, k0, K, half, idx16);
    f.al[i] = load_fragA(Alo, m0 + 16 * i, k0, K, half, idx16);
  }
#pragma unroll
  for (int j = 0; j < 2; ++j) {
    f.bh[j] = load_fragB(Bhi, n0 + 16 * j, k0, K, half, idx16);
    f.bl[j] = load_fragB(Blo, n0 + 16 * j, k0, K, half, idx16);
  }
}

__device__ __forceinline__ void do_wmma(v8f acc[4][2], const Frags& f) {
#pragma unroll
  for (int i = 0; i < 4; ++i)
#pragma unroll
    for (int j = 0; j < 2; ++j) {
      acc[i][j] = wmma_bf16(f.ah[i], f.bh[j], acc[i][j]);  // hi*hi
      acc[i][j] = wmma_bf16(f.ah[i], f.bl[j], acc[i][j]);  // hi*lo
      acc[i][j] = wmma_bf16(f.al[i], f.bh[j], acc[i][j]);  // lo*hi
    }
}

// ---- split-bf16 WMMA GEMM: C[M,N] = A[M,K] @ Bt[N,K]^T ---------------------
// block = 128 threads = 4 waves in a 2x2 grid; each wave owns a 64x32 C tile
// (4 M-frags x 2 N-frags) -> block computes 128x64 of C.
// k-loop unrolled by 2 with ping-pong fragment buffers (K % 64 == 0 always).
__global__ __launch_bounds__(128) void k_gemm(const unsigned short* __restrict__ Ahi,
                                              const unsigned short* __restrict__ Alo,
                                              const unsigned short* __restrict__ Bhi,
                                              const unsigned short* __restrict__ Blo,
                                              float* __restrict__ C,
                                              int M, int N, int K) {
  const int lane  = threadIdx.x & 31;
  const int w     = threadIdx.x >> 5;
  const int half  = lane >> 4;
  const int idx16 = lane & 15;
  const int m0 = blockIdx.y * 128 + (w >> 1) * 64;
  const int n0 = blockIdx.x * 64  + (w & 1) * 32;

  v8f acc[4][2];
#pragma unroll
  for (int i = 0; i < 4; ++i)
#pragma unroll
    for (int j = 0; j < 2; ++j)
      acc[i][j] = (v8f){0.f, 0.f, 0.f, 0.f, 0.f, 0.f, 0.f, 0.f};

  Frags fA, fB;
  load_frags(fA, Ahi, Alo, Bhi, Blo, m0, n0, 0, K, half, idx16);

  for (int k0 = 0; k0 < K; k0 += 64) {
    load_frags(fB, Ahi, Alo, Bhi, Blo, m0, n0, k0 + 32, K, half, idx16);
    do_wmma(acc, fA);
    if (k0 + 64 < K)
      load_frags(fA, Ahi, Alo, Bhi, Blo, m0, n0, k0 + 64, K, half, idx16);
    do_wmma(acc, fB);
  }

  // C/D layout: lane idx16 = column, VGPR v -> row v + half*8
#pragma unroll
  for (int i = 0; i < 4; ++i)
#pragma unroll
    for (int j = 0; j < 2; ++j) {
      int n = n0 + 16 * j + idx16;
      int r = m0 + 16 * i + half * 8;
#pragma unroll
      for (int v = 0; v < 8; ++v)
        C[(size_t)(r + v) * N + n] = acc[i][j][v];
    }
}

// ---------------------------------------------------------------------------
extern "C" void kernel_launch(void* const* d_in, const int* in_sizes, int n_in,
                              void* d_out, int out_size, void* d_ws, size_t ws_size,
                              hipStream_t stream) {
  const float* feats = (const float*)d_in[0];
  const int*   src   = (const int*)d_in[1];
  const int*   dst   = (const int*)d_in[2];
  const float* W0    = (const float*)d_in[3];
  const float* b0    = (const float*)d_in[4];
  const float* W1    = (const float*)d_in[5];
  const float* b1    = (const float*)d_in[6];
  float* out = (float*)d_out;

  char* ws = (char*)d_ws;
  auto alloc = [&](size_t bytes) {
    char* p = ws;
    ws += (bytes + 255) & ~(size_t)255;
    return p;
  };

  unsigned short* feats_hi = (unsigned short*)alloc((size_t)NN * IN_DIM * 2);
  unsigned short* feats_lo = (unsigned short*)alloc((size_t)NN * IN_DIM * 2);
  unsigned short* W0t_hi   = (unsigned short*)alloc((size_t)H0D * IN_DIM * 2);
  unsigned short* W0t_lo   = (unsigned short*)alloc((size_t)H0D * IN_DIM * 2);
  unsigned short* W1t_hi   = (unsigned short*)alloc((size_t)H1D * H0D * 2);
  unsigned short* W1t_lo   = (unsigned short*)alloc((size_t)H1D * H0D * 2);
  float* Y0   = (float*)alloc((size_t)NN * H0D * 4);
  float* Agg0 = (float*)alloc((size_t)NN * H0D * 4);
  unsigned short* H_hi = (unsigned short*)alloc((size_t)NN * H0D * 2);
  unsigned short* H_lo = (unsigned short*)alloc((size_t)NN * H0D * 2);
  float* Y1   = (float*)alloc((size_t)NN * H1D * 4);
  float* Agg1 = (float*)alloc((size_t)NN * H1D * 4);
  unsigned short* Z_hi = (unsigned short*)alloc((size_t)NN * H1D * 2);
  unsigned short* Z_lo = (unsigned short*)alloc((size_t)NN * H1D * 2);

  // 1) split fp32 -> bf16 hi/lo (feats) and transposed weights
  k_split  <<<(NN * IN_DIM + 255) / 256, 256, 0, stream>>>(feats, feats_hi, feats_lo, NN * IN_DIM);
  k_split_t<<<(IN_DIM * H0D + 255) / 256, 256, 0, stream>>>(W0, W0t_hi, W0t_lo, IN_DIM, H0D);
  k_split_t<<<(H0D * H1D + 255) / 256, 256, 0, stream>>>(W1, W1t_hi, W1t_lo, H0D, H1D);

  // 2) Y0 = feats @ W0     [8192,512]x[512,256]
  {
    dim3 g(H0D / 64, NN / 128);
    k_gemm<<<g, 128, 0, stream>>>(feats_hi, feats_lo, W0t_hi, W0t_lo, Y0, NN, H0D, IN_DIM);
  }

  // 3) Agg0 = segment_sum(Y0[src] -> dst)   (agg commutes with linear layer)
  k_zero   <<<(NN * H0D + 255) / 256, 256, 0, stream>>>(Agg0, NN * H0D);
  k_scatter<<<(EE * (H0D / 4) + 255) / 256, 256, 0, stream>>>(Y0, src, dst, Agg0, H0D, EE);

  // 4) H = relu(Agg0 + b0) -> bf16 hi/lo
  k_bias_split<<<(NN * H0D + 255) / 256, 256, 0, stream>>>(Agg0, b0, H_hi, H_lo, H0D, NN * H0D, 1);

  // 5) Y1 = H @ W1          [8192,256]x[256,128]
  {
    dim3 g(H1D / 64, NN / 128);
    k_gemm<<<g, 128, 0, stream>>>(H_hi, H_lo, W1t_hi, W1t_lo, Y1, NN, H1D, H0D);
  }

  // 6) Agg1 = segment_sum(Y1[src] -> dst)
  k_zero   <<<(NN * H1D + 255) / 256, 256, 0, stream>>>(Agg1, NN * H1D);
  k_scatter<<<(EE * (H1D / 4) + 255) / 256, 256, 0, stream>>>(Y1, src, dst, Agg1, H1D, EE);

  // 7) Z = Agg1 + b1 -> bf16 hi/lo
  k_bias_split<<<(NN * H1D + 255) / 256, 256, 0, stream>>>(Agg1, b1, Z_hi, Z_lo, H1D, NN * H1D, 0);

  // 8) out = Z @ Z^T        [8192,128]x[128,8192] -> 256MB fp32
  {
    dim3 g(NN / 64, NN / 128);
    k_gemm<<<g, 128, 0, stream>>>(Z_hi, Z_lo, Z_hi, Z_lo, out, NN, NN, H1D);
  }
}
